// MixtureOfRecursionsEncoder_19104014532936
// MI455X (gfx1250) — compile-verified, matrix operand order checked by reading, not run on hardware
//
#include <hip/hip_runtime.h>
#include <hip/hip_bf16.h>
#include <math.h>

#define DEV __device__ __forceinline__

typedef __attribute__((ext_vector_type(16))) _Float16 v16h;
typedef __attribute__((ext_vector_type(8)))  _Float16 v8h;
typedef __attribute__((ext_vector_type(8)))  float    v8f;
typedef __attribute__((ext_vector_type(4)))  int      v4i;

namespace {
constexpr int Bc = 8, Lc = 1024, Dc = 1024, Hc = 16, MDc = 6, DFc = 4096;
constexpr int Tok = Bc * Lc;  // 8192 token rows
}

// CDNA5 async global->LDS path (ASYNCcnt-tracked), with safe fallback.
#if defined(__HIP_DEVICE_COMPILE__) && __has_builtin(__builtin_amdgcn_global_load_async_to_lds_b128)
#define ATHENA_ASYNC_LDS 1
#endif

#ifdef ATHENA_ASYNC_LDS
typedef __attribute__((address_space(1))) v4i gas_v4i;   // global (HIP __device__)
typedef __attribute__((address_space(3))) v4i las_v4i;   // LDS
#endif

DEV void cp16_async(const _Float16* g, _Float16* l) {
#ifdef ATHENA_ASYNC_LDS
  __builtin_amdgcn_global_load_async_to_lds_b128(
      (gas_v4i*)const_cast<_Float16*>(g), (las_v4i*)l, 0, 0);
#else
  *(v8h*)l = *(const v8h*)g;
#endif
}

template <int N>
DEV void wait_async() {
#ifdef ATHENA_ASYNC_LDS
#if __has_builtin(__builtin_amdgcn_s_wait_asynccnt)
  __builtin_amdgcn_s_wait_asynccnt(N);
#else
  if constexpr (N == 0) asm volatile("s_wait_asynccnt 0" ::: "memory");
  else                  asm volatile("s_wait_asynccnt 8" ::: "memory");
#endif
#endif
}

DEV v8f v8f_zero() {
  v8f z;
#pragma unroll
  for (int i = 0; i < 8; ++i) z[i] = 0.f;
  return z;
}

// Load a 16x32 f16 A/B-style WMMA fragment from a row-major (row, K) array.
// Per CDNA5 ISA 7.12.2: lane L -> row L%16; halfs 0..7 -> K = 8*(L>=16)+0..7,
// halfs 8..15 -> K = 16+8*(L>=16)+0..7.  Two 16B loads per lane.
DEV v16h frag_ld(const _Float16* base, int stride, int row) {
  const int lane = threadIdx.x & 31;
  const int l16 = lane & 15, hi = lane >> 4;
  const _Float16* p = base + (size_t)(row + l16) * stride + hi * 8;
  v8h a = *(const v8h*)p;
  v8h b = *(const v8h*)(p + 16);
  v16h r;
#pragma unroll
  for (int i = 0; i < 8; ++i) { r[i] = a[i]; r[i + 8] = b[i]; }
  return r;
}

DEV float red_max16(float v) {
#pragma unroll
  for (int m = 1; m <= 8; m <<= 1) v = fmaxf(v, __shfl_xor(v, m, 32));
  return v;
}
DEV float red_sum16(float v) {
#pragma unroll
  for (int m = 1; m <= 8; m <<= 1) v += __shfl_xor(v, m, 32);
  return v;
}

DEV float gelu_exact(float v) {
  return 0.5f * v * (1.f + erff(v * 0.70710678118654752f));
}

// ---------------------------------------------------------------------------
// Elementwise helpers
// ---------------------------------------------------------------------------
__global__ void copy_f32(const float* __restrict__ src, float* __restrict__ dst, size_t n) {
  size_t i = (size_t)blockIdx.x * blockDim.x + threadIdx.x;
  size_t stride = (size_t)gridDim.x * blockDim.x;
  for (; i < n; i += stride) dst[i] = src[i];
}

__global__ void cvt_f32_f16(const float* __restrict__ src, _Float16* __restrict__ dst, size_t n) {
  size_t i = (size_t)blockIdx.x * blockDim.x + threadIdx.x;
  size_t stride = (size_t)gridDim.x * blockDim.x;
  for (; i < n; i += stride) dst[i] = (_Float16)src[i];
}

// ---------------------------------------------------------------------------
// LayerNorm (f32 in, f16 out).  One 256-thread block per row of D=1024.
// ---------------------------------------------------------------------------
__global__ __launch_bounds__(256)
void layernorm_kernel(const float* __restrict__ x, const float* __restrict__ gw,
                      const float* __restrict__ gb, _Float16* __restrict__ out) {
  const int row = blockIdx.x;
  const int tid = threadIdx.x;
  const float4 xv = ((const float4*)(x + (size_t)row * Dc))[tid];
  float s = xv.x + xv.y + xv.z + xv.w;
#pragma unroll
  for (int m = 1; m <= 16; m <<= 1) s += __shfl_xor(s, m, 32);
  __shared__ float r1[8], r2[8];
  const int lane = tid & 31, wave = tid >> 5;
  if (lane == 0) r1[wave] = s;
  __syncthreads();
  float tot = 0.f;
#pragma unroll
  for (int i = 0; i < 8; ++i) tot += r1[i];
  const float mu = tot * (1.f / Dc);
  const float dx = xv.x - mu, dy = xv.y - mu, dz = xv.z - mu, dw = xv.w - mu;
  float sq = dx * dx + dy * dy + dz * dz + dw * dw;
#pragma unroll
  for (int m = 1; m <= 16; m <<= 1) sq += __shfl_xor(sq, m, 32);
  if (lane == 0) r2[wave] = sq;
  __syncthreads();
  float vtot = 0.f;
#pragma unroll
  for (int i = 0; i < 8; ++i) vtot += r2[i];
  const float rstd = rsqrtf(vtot * (1.f / Dc) + 1e-5f);
  const float4 wv = ((const float4*)gw)[tid];
  const float4 bv = ((const float4*)gb)[tid];
  _Float16* orow = out + (size_t)row * Dc + tid * 4;
  orow[0] = (_Float16)(dx * rstd * wv.x + bv.x);
  orow[1] = (_Float16)(dy * rstd * wv.y + bv.y);
  orow[2] = (_Float16)(dz * rstd * wv.z + bv.z);
  orow[3] = (_Float16)(dw * rstd * wv.w + bv.w);
}

// ---------------------------------------------------------------------------
// WMMA GEMM: C[M,N] = A[M,K](f16) * W[N,K]^T(f16) + bias, fused epilogue.
//   MODE 0: store f16            (QKV projection)
//   MODE 1: GELU -> f16          (FFN1, router hidden)
//   MODE 2: res_out = res_in + C (attn out-proj -> x1, f32)
//   MODE 3: if depth>=level: res_out = res_in + C   (FFN2 masked residual)
// 128x128 C-tile / 256 threads; 64-deep K-steps, double-buffered LDS stages
// filled by async direct-to-LDS loads (8 in flight per wave).
// ---------------------------------------------------------------------------
template <int MODE>
__global__ __launch_bounds__(256, 2)
void gemm_wmma(const _Float16* __restrict__ A, const _Float16* __restrict__ W,
               const float* __restrict__ bias, int K, int N,
               const float* __restrict__ res_in, float* __restrict__ res_out,
               _Float16* __restrict__ out_h,
               const int* __restrict__ depths, int level) {
  __shared__ _Float16 sA[2][128 * 80];
  __shared__ _Float16 sW[2][128 * 80];
  const int tid = threadIdx.x;
  const int wave = tid >> 5;
  const int waveM = wave >> 2, waveN = wave & 3;
  const int wgN = blockIdx.x * 128, wgM = blockIdx.y * 128;

  auto stage = [&](int buf, int k0) {
#pragma unroll
    for (int t = 0; t < 4; ++t) {
      const int cidx = tid + t * 256;           // 0..1023
      const int row = cidx >> 3, cc = cidx & 7; // 128 rows x 8 chunks of 16B
      cp16_async(&A[(size_t)(wgM + row) * K + k0 + cc * 8], &sA[buf][row * 80 + cc * 8]);
      cp16_async(&W[(size_t)(wgN + row) * K + k0 + cc * 8], &sW[buf][row * 80 + cc * 8]);
#ifndef ATHENA_ASYNC_LDS
      if (k0 + 64 < K) {
        __builtin_prefetch(&A[(size_t)(wgM + row) * K + k0 + 64 + cc * 8], 0, 0);
        __builtin_prefetch(&W[(size_t)(wgN + row) * K + k0 + 64 + cc * 8], 0, 0);
      }
#endif
    }
  };

  v8f c[4][2];
#pragma unroll
  for (int i = 0; i < 4; ++i)
#pragma unroll
    for (int j = 0; j < 2; ++j) c[i][j] = v8f_zero();

  stage(0, 0);
  const int ksteps = K >> 6;
  for (int s = 0; s < ksteps; ++s) {
    const int cur = s & 1;
    if (s + 1 < ksteps) {
      stage(cur ^ 1, (s + 1) << 6);  // issue next stage (8 async/wave in flight)
      wait_async<8>();               // current stage resident, next still flying
    } else {
      wait_async<0>();
    }
    __syncthreads();
    const _Float16* pA = sA[cur];
    const _Float16* pW = sW[cur];
#pragma unroll
    for (int kb = 0; kb < 2; ++kb) {  // two 32-deep sub-steps of the 64-K tile
      const v16h wf0 = frag_ld(pW + kb * 32, 80, waveN * 32);
      const v16h wf1 = frag_ld(pW + kb * 32, 80, waveN * 32 + 16);
#pragma unroll
      for (int i = 0; i < 4; ++i) {
        const v16h af = frag_ld(pA + kb * 32, 80, waveM * 64 + i * 16);
        c[i][0] = __builtin_amdgcn_wmma_f32_16x16x32_f16(false, af, false, wf0,
                                                         (short)0, c[i][0], false, false);
        c[i][1] = __builtin_amdgcn_wmma_f32_16x16x32_f16(false, af, false, wf1,
                                                         (short)0, c[i][1], false, false);
      }
    }
    __syncthreads();  // protects buffer reuse two stages later
  }

  const int lane = tid & 31, l16 = lane & 15, hi = lane >> 4;
#pragma unroll
  for (int i = 0; i < 4; ++i)
#pragma unroll
    for (int j = 0; j < 2; ++j)
#pragma unroll
      for (int r = 0; r < 8; ++r) {
        const int gr = wgM + waveM * 64 + i * 16 + r + 8 * hi;
        const int gc = wgN + waveN * 32 + j * 16 + l16;
        const float v = c[i][j][r] + bias[gc];
        if constexpr (MODE == 0) {
          out_h[(size_t)gr * N + gc] = (_Float16)v;
        } else if constexpr (MODE == 1) {
          out_h[(size_t)gr * N + gc] = (_Float16)gelu_exact(v);
        } else if constexpr (MODE == 2) {
          const size_t idx = (size_t)gr * N + gc;
          res_out[idx] = res_in[idx] + v;
        } else {
          const size_t idx = (size_t)gr * N + gc;
          if (depths[gr] >= level) res_out[idx] = res_in[idx] + v;
        }
      }
}

// ---------------------------------------------------------------------------
// Flash attention.  grid = (L/128, B*H); 8 waves x 16 query rows per WG.
// 64-key tiles: 16 WMMAs per tile, halved softmax-butterfly & barrier cost.
// qkv layout: [b, l, 3*D] with q|k|v at offsets 0|D|2D, head h at h*64.
// ---------------------------------------------------------------------------
__global__ __launch_bounds__(256, 2)
void attn_kernel(const _Float16* __restrict__ qkv, _Float16* __restrict__ ctx) {
  __shared__ _Float16 sK[64 * 80];      // [key][head-dim]
  __shared__ _Float16 sVT[64 * 80];     // [head-dim][key]
  __shared__ _Float16 sP[8][16 * 64];   // per-wave P transpose staging

  const int tid = threadIdx.x, wave = tid >> 5, lane = tid & 31;
  const int l16 = lane & 15, hi = lane >> 4;
  const int b = blockIdx.y >> 4, h = blockIdx.y & 15;
  const int q0 = blockIdx.x * 128 + wave * 16;
  const size_t tokbase = (size_t)b * Lc;
  const _Float16* qbase = qkv + tokbase * 3072 + h * 64;

  const v16h qf0 = frag_ld(qbase, 3072, q0);        // head dims 0..31
  const v16h qf1 = frag_ld(qbase + 32, 3072, q0);   // head dims 32..63

  float mrow[8], lrow[8];
  v8f o[4];
#pragma unroll
  for (int r = 0; r < 8; ++r) { mrow[r] = -3.0e38f; lrow[r] = 0.f; }
#pragma unroll
  for (int n = 0; n < 4; ++n) o[n] = v8f_zero();

  for (int kk = 0; kk < Lc; kk += 64) {
    {  // stage K (async, row-major) and V (transposed scatter) tiles
#pragma unroll
      for (int t = 0; t < 2; ++t) {
        const int cidx = tid + t * 256;            // 0..511
        const int r = cidx >> 3, cc = cidx & 7;    // 64 rows x 8 chunks
        const _Float16* kp = qkv + (tokbase + kk + r) * 3072 + Dc + h * 64 + cc * 8;
        cp16_async(kp, &sK[r * 80 + cc * 8]);
        const _Float16* vp = kp + Dc;
        const v8h vv = *(const v8h*)vp;
#pragma unroll
        for (int i = 0; i < 8; ++i) sVT[(cc * 8 + i) * 80 + r] = vv[i];
        if (kk + 64 < Lc) {
          __builtin_prefetch(kp + 64 * 3072, 0, 0);
          __builtin_prefetch(vp + 64 * 3072, 0, 0);
        }
      }
    }
    wait_async<0>();
    __syncthreads();

    v8f s[4];
#pragma unroll
    for (int n = 0; n < 4; ++n) {                  // keys n*16..n*16+15
      s[n] = v8f_zero();
      const v16h kfa = frag_ld(sK, 80, n * 16);    // dims 0..31
      s[n] = __builtin_amdgcn_wmma_f32_16x16x32_f16(false, qf0, false, kfa, (short)0, s[n], false, false);
      const v16h kfb = frag_ld(sK + 32, 80, n * 16);  // dims 32..63
      s[n] = __builtin_amdgcn_wmma_f32_16x16x32_f16(false, qf1, false, kfb, (short)0, s[n], false, false);
    }

    _Float16* sPw = sP[wave];
#pragma unroll
    for (int r = 0; r < 8; ++r) {
      const float a0 = s[0][r] * 0.125f, a1 = s[1][r] * 0.125f;   // 1/sqrt(64)
      const float a2 = s[2][r] * 0.125f, a3 = s[3][r] * 0.125f;
      const float rm = red_max16(fmaxf(fmaxf(a0, a1), fmaxf(a2, a3)));
      const float mnew = fmaxf(mrow[r], rm);
      const float corr = __expf(mrow[r] - mnew);
      const float p0 = __expf(a0 - mnew), p1 = __expf(a1 - mnew);
      const float p2 = __expf(a2 - mnew), p3 = __expf(a3 - mnew);
      lrow[r] = lrow[r] * corr + red_sum16(p0 + p1 + p2 + p3);
      mrow[r] = mnew;
#pragma unroll
      for (int n = 0; n < 4; ++n) o[n][r] *= corr;
      const int prow = (r + 8 * hi) * 64;
      sPw[prow + l16] = (_Float16)p0;
      sPw[prow + 16 + l16] = (_Float16)p1;
      sPw[prow + 32 + l16] = (_Float16)p2;
      sPw[prow + 48 + l16] = (_Float16)p3;
    }

    const v16h pf0 = frag_ld(sPw, 64, 0);        // q-rows x keys 0..31
    const v16h pf1 = frag_ld(sPw + 32, 64, 0);   // q-rows x keys 32..63
#pragma unroll
    for (int nd = 0; nd < 4; ++nd) {             // head-dim tiles of 16
      const v16h vf0 = frag_ld(sVT, 80, nd * 16);
      o[nd] = __builtin_amdgcn_wmma_f32_16x16x32_f16(false, pf0, false, vf0, (short)0, o[nd], false, false);
      const v16h vf1 = frag_ld(sVT + 32, 80, nd * 16);
      o[nd] = __builtin_amdgcn_wmma_f32_16x16x32_f16(false, pf1, false, vf1, (short)0, o[nd], false, false);
    }
    __syncthreads();
  }

#pragma unroll
  for (int n = 0; n < 4; ++n)
#pragma unroll
    for (int r = 0; r < 8; ++r) {
      const size_t tok = tokbase + q0 + r + 8 * hi;
      ctx[tok * Dc + h * 64 + n * 16 + l16] = (_Float16)(o[n][r] / lrow[r]);
    }
}

// ---------------------------------------------------------------------------
// Router tail: logits = h @ rt_w2^T + b2 (N=6), argmax -> depths.
// ---------------------------------------------------------------------------
__global__ __launch_bounds__(256)
void router_kernel(const _Float16* __restrict__ hbuf, const float* __restrict__ w2,
                   const float* __restrict__ b2, float* __restrict__ logitsf,
                   float* __restrict__ depthsf, int* __restrict__ depthsi) {
  const int row = blockIdx.x;
  const int tid = threadIdx.x;
  float acc[MDc];
#pragma unroll
  for (int m = 0; m < MDc; ++m) acc[m] = 0.f;
  const _Float16* hr = hbuf + (size_t)row * Dc;
  for (int i = tid; i < Dc; i += 256) {
    const float hv = (float)hr[i];
#pragma unroll
    for (int m = 0; m < MDc; ++m) acc[m] += hv * w2[m * Dc + i];
  }
  __shared__ float red[8][MDc];
  const int lane = tid & 31, wave = tid >> 5;
#pragma unroll
  for (int m = 0; m < MDc; ++m)
#pragma unroll
    for (int s = 1; s <= 16; s <<= 1) acc[m] += __shfl_xor(acc[m], s, 32);
  if (lane == 0)
#pragma unroll
    for (int m = 0; m < MDc; ++m) red[wave][m] = acc[m];
  __syncthreads();
  if (tid == 0) {
    float best = -3.0e38f;
    int bi = 0;
#pragma unroll
    for (int m = 0; m < MDc; ++m) {
      float v = b2[m];
      for (int wv = 0; wv < 8; ++wv) v += red[wv][m];
      logitsf[(size_t)row * MDc + m] = v;
      if (v > best) { best = v; bi = m; }  // strict >: first max, jnp.argmax
    }
    depthsi[row] = bi + 1;
    depthsf[row] = (float)(bi + 1);
  }
}

// ---------------------------------------------------------------------------
// Host orchestration
// ---------------------------------------------------------------------------
extern "C" void kernel_launch(void* const* d_in, const int* in_sizes, int n_in,
                              void* d_out, int out_size, void* d_ws, size_t ws_size,
                              hipStream_t stream) {
  (void)in_sizes; (void)n_in; (void)out_size; (void)ws_size;
  const float* x           = (const float*)d_in[0];
  const float* attn_norm_w = (const float*)d_in[1];
  const float* attn_norm_b = (const float*)d_in[2];
  const float* in_proj_w   = (const float*)d_in[3];
  const float* in_proj_b   = (const float*)d_in[4];
  const float* out_proj_w  = (const float*)d_in[5];
  const float* out_proj_b  = (const float*)d_in[6];
  const float* ffn_norm_w  = (const float*)d_in[7];
  const float* ffn_norm_b  = (const float*)d_in[8];
  const float* ffn_w1      = (const float*)d_in[9];
  const float* ffn_b1      = (const float*)d_in[10];
  const float* ffn_w2      = (const float*)d_in[11];
  const float* ffn_b2      = (const float*)d_in[12];
  const float* rt_norm_w   = (const float*)d_in[13];
  const float* rt_norm_b   = (const float*)d_in[14];
  const float* rt_w1       = (const float*)d_in[15];
  const float* rt_b1       = (const float*)d_in[16];
  const float* rt_w2       = (const float*)d_in[17];
  const float* rt_b2       = (const float*)d_in[18];

  // d_out = [cur (B*L*D) | depths (B*L) | logits (B*L*MD)] as f32
  float* xcur    = (float*)d_out;
  float* depthsf = xcur + (size_t)Tok * Dc;
  float* logitsf = depthsf + Tok;

  // workspace carve-up
  char* wp = (char*)d_ws;
  auto alloc = [&](size_t bytes) {
    char* p = wp;
    wp += (bytes + 255) & ~(size_t)255;
    return p;
  };
  float*    x1   = (float*)alloc((size_t)Tok * Dc * 4);
  _Float16* ln   = (_Float16*)alloc((size_t)Tok * Dc * 2);
  _Float16* qkvb = (_Float16*)alloc((size_t)Tok * 3 * Dc * 2);
  _Float16* ctxb = (_Float16*)alloc((size_t)Tok * Dc * 2);
  _Float16* hb   = (_Float16*)alloc((size_t)Tok * DFc * 2);
  _Float16* w_in = (_Float16*)alloc((size_t)3 * Dc * Dc * 2);
  _Float16* w_o  = (_Float16*)alloc((size_t)Dc * Dc * 2);
  _Float16* w_f1 = (_Float16*)alloc((size_t)DFc * Dc * 2);
  _Float16* w_f2 = (_Float16*)alloc((size_t)Dc * DFc * 2);
  _Float16* w_r1 = (_Float16*)alloc((size_t)Dc * Dc * 2);
  int*      depthsi = (int*)alloc((size_t)Tok * 4);

  // --- one-time per-launch prep: f32 -> f16 weights, x -> xcur
  cvt_f32_f16<<<2048, 256, 0, stream>>>(in_proj_w, w_in, (size_t)3 * Dc * Dc);
  cvt_f32_f16<<<2048, 256, 0, stream>>>(out_proj_w, w_o, (size_t)Dc * Dc);
  cvt_f32_f16<<<2048, 256, 0, stream>>>(ffn_w1, w_f1, (size_t)DFc * Dc);
  cvt_f32_f16<<<2048, 256, 0, stream>>>(ffn_w2, w_f2, (size_t)Dc * DFc);
  cvt_f32_f16<<<2048, 256, 0, stream>>>(rt_w1, w_r1, (size_t)Dc * Dc);
  copy_f32<<<4096, 256, 0, stream>>>(x, xcur, (size_t)Tok * Dc);

  // --- router: depths/logits
  layernorm_kernel<<<Tok, 256, 0, stream>>>(x, rt_norm_w, rt_norm_b, ln);
  gemm_wmma<1><<<dim3(Dc / 128, Tok / 128), 256, 0, stream>>>(
      ln, w_r1, rt_b1, Dc, Dc, nullptr, nullptr, hb, nullptr, 0);
  router_kernel<<<Tok, 256, 0, stream>>>(hb, rt_w2, rt_b2, logitsf, depthsf, depthsi);

  // --- recursion levels
  for (int level = 1; level <= MDc; ++level) {
    layernorm_kernel<<<Tok, 256, 0, stream>>>(xcur, attn_norm_w, attn_norm_b, ln);
    gemm_wmma<0><<<dim3(3 * Dc / 128, Tok / 128), 256, 0, stream>>>(
        ln, w_in, in_proj_b, Dc, 3 * Dc, nullptr, nullptr, qkvb, nullptr, 0);
    attn_kernel<<<dim3(Lc / 128, Bc * Hc), 256, 0, stream>>>(qkvb, ctxb);
    gemm_wmma<2><<<dim3(Dc / 128, Tok / 128), 256, 0, stream>>>(
        ctxb, w_o, out_proj_b, Dc, Dc, xcur, x1, nullptr, nullptr, 0);
    layernorm_kernel<<<Tok, 256, 0, stream>>>(x1, ffn_norm_w, ffn_norm_b, ln);
    gemm_wmma<1><<<dim3(DFc / 128, Tok / 128), 256, 0, stream>>>(
        ln, w_f1, ffn_b1, Dc, DFc, nullptr, nullptr, hb, nullptr, 0);
    gemm_wmma<3><<<dim3(Dc / 128, Tok / 128), 256, 0, stream>>>(
        hb, w_f2, ffn_b2, DFc, Dc, x1, xcur, nullptr, depthsi, level);
  }
}
